// Model_36696200577168
// MI455X (gfx1250) — matrix operand order repaired and checked by reading, not compiled, not run on hardware
//
#include <hip/hip_runtime.h>
#include <hip/hip_bf16.h>

// ---------------- problem constants ----------------
#define B_    16
#define P_    128
#define M_    2048
#define NN_   32
#define K_    8
#define LAYERS_ 10
#define C_    64
#define ROWS  (B_ * M_)          // 32768
#define NCOLS 576                // 64 (W2) + 8*64 (Wk)
#define GCOLS 512                // K_*C_ columns kept in g buffer
#define BN_EPS 1e-5f

typedef _Float16 half8  __attribute__((ext_vector_type(8)));
typedef _Float16 half16 __attribute__((ext_vector_type(16)));
typedef float    float8 __attribute__((ext_vector_type(8)));

// ---------------- fragment loaders (CDNA5 WMMA f16 layouts) ----------------
// A 16x32 f16: lanes 0-15 -> row, k halves {k0..k0+7, k0+16..k0+23};
//              lanes 16-31 -> same rows, +8 on both k groups.
__device__ __forceinline__ half16 load_fragA(const _Float16* __restrict__ base,
                                             int ld, int row0, int k0, int lane) {
  int r = row0 + (lane & 15);
  int kb = k0 + ((lane >> 4) << 3);            // +8 for upper half-wave
  const _Float16* p = base + (size_t)r * ld + kb;
  half8 lo = *(const half8*)(p);               // k = kb .. kb+7
  half8 hi = *(const half8*)(p + 16);          // k = kb+16 .. kb+23
  half16 v;
#pragma unroll
  for (int i = 0; i < 8; ++i) { v[i] = lo[i]; v[i + 8] = hi[i]; }
  return v;
}

// B 32x16 f16: lanes 0-15 -> col, k halves {k0..k0+7, k0+16..k0+23}; lanes 16-31 -> +8 on both k groups.
__device__ __forceinline__ half16 load_fragB(const _Float16* __restrict__ base,
                                             int ld, int col0, int k0, int lane) {
  int c = col0 + (lane & 15);
  const _Float16* p = base + (size_t)c * ld + k0 + ((lane >> 4) << 3);
  half8 lo = *(const half8*)(p);
  half8 hi = *(const half8*)(p + 16);
  half16 v;
#pragma unroll
  for (int i = 0; i < 8; ++i) { v[i] = lo[i]; v[i + 8] = hi[i]; }
  return v;
}

// ---------------- kernels ----------------

// x (B,P,M) fp32 -> h (B,M,P) f16 : LDS-tiled transpose, coalesced both sides
__global__ void k_transpose_x(const float* __restrict__ x, _Float16* __restrict__ h) {
  __shared__ float tile[32][33];
  int m0 = blockIdx.x * 32, p0 = blockIdx.y * 32, b = blockIdx.z;
  int tx = threadIdx.x, ty = threadIdx.y;                  // (32, 8)
#pragma unroll
  for (int i = ty; i < 32; i += 8)
    tile[i][tx] = x[(size_t)b * P_ * M_ + (size_t)(p0 + i) * M_ + m0 + tx];
  __syncthreads();
#pragma unroll
  for (int i = ty; i < 32; i += 8)
    h[(size_t)b * M_ * P_ + (size_t)(m0 + i) * P_ + p0 + tx] = (_Float16)tile[tx][i];
}

// attn[k][m][n] = softmax_n( exp(-0.5 * sum_f sigma[k,f]*(maps[e,f]-mu[k,f])^2) )
__global__ void k_attn(const float* __restrict__ maps, const float* __restrict__ sigma,
                       const float* __restrict__ mu, float* __restrict__ attn) {
  int wave = (blockIdx.x * blockDim.x + threadIdx.x) >> 5;   // K_*M_ waves
  int lane = threadIdx.x & 31;                                // n
  int k = wave >> 11;          // / M_
  int m = wave & (M_ - 1);
  int e = m * NN_ + lane;
  float q = 0.f;
#pragma unroll
  for (int f = 0; f < 5; ++f) {
    float u = maps[(size_t)e * 5 + f] - mu[k * 5 + f];
    q += u * u * sigma[k * 5 + f];
  }
  float logit = __expf(-0.5f * q);
  float mx = logit;
#pragma unroll
  for (int off = 16; off >= 1; off >>= 1) mx = fmaxf(mx, __shfl_xor(mx, off, 32));
  float ex = __expf(logit - mx);
  float sum = ex;
#pragma unroll
  for (int off = 16; off >= 1; off >>= 1) sum += __shfl_xor(sum, off, 32);
  attn[(size_t)k * M_ * NN_ + (size_t)m * NN_ + lane] = ex / sum;
}

// Pack [W2 | Wk0..Wk7] transposed into f16 (NCOLS x din) + biasSum = b2 + sum_k bk
__global__ void k_wconv(const float* __restrict__ W2, const float* __restrict__ Wk,
                        const float* __restrict__ b2, const float* __restrict__ bk,
                        _Float16* __restrict__ Wt, float* __restrict__ biasSum, int din) {
  int t = blockIdx.x * blockDim.x + threadIdx.x;
  int nw = NCOLS * din;
  if (t < nw) {
    int n = t / din, k = t % din;
    float v;
    if (n < C_) v = W2[(size_t)k * C_ + n];
    else {
      int kb = (n - C_) >> 6, c = (n - C_) & 63;
      v = Wk[(size_t)kb * din * C_ + (size_t)k * C_ + c];
    }
    Wt[(size_t)n * din + k] = (_Float16)v;
  } else if (t < nw + C_) {
    int c = t - nw;
    float s = b2[c];
#pragma unroll
    for (int k = 0; k < K_; ++k) s += bk[k * C_ + c];
    biasSum[c] = s;
  }
}

// fc1_w (C_,512) -> Wt (512 x C_) f16
__global__ void k_wconv_fc1(const float* __restrict__ w, _Float16* __restrict__ Wt) {
  int t = blockIdx.x * blockDim.x + threadIdx.x;
  if (t >= GCOLS * C_) return;
  int n = t >> 6, k = t & 63;
  Wt[(size_t)n * C_ + k] = (_Float16)w[(size_t)k * GCOLS + n];
}

// WMMA GEMM: A (ROWS x din) f16 row-major, Bt (ncols x din) f16 row-major.
// Each wave computes a 16x64 tile (4 accumulators sharing one A fragment).
// EPI 0: col<64  -> h2 = acc + biasSum[col] (fp32)
//        col>=64 -> g swizzled f16: g[row*512 + c*8 + kidx], kidx=(col-64)/64
// EPI 1: g = relu(acc + bias[col]) f16, linear layout   (fc1)
template <int EPI>
__global__ void k_gemm_wmma(const _Float16* __restrict__ A, const _Float16* __restrict__ Bt,
                            int din, int ncols, int ksteps, const float* __restrict__ bias,
                            float* __restrict__ h2, _Float16* __restrict__ g) {
  int wave = (blockIdx.x * blockDim.x + threadIdx.x) >> 5;
  int lane = threadIdx.x & 31;
  int ntg = ncols >> 6;                  // groups of 4 N-tiles: 9 (layer) / 8 (fc1)
  int tm = wave / ntg;
  int tg = wave - tm * ntg;
  if (tm >= (ROWS >> 4)) return;         // grid exact; never taken
  float8 acc[4] = {{}, {}, {}, {}};
  for (int kk = 0; kk < ksteps; ++kk) {
    half16 a = load_fragA(A, din, tm << 4, kk << 5, lane);
#pragma unroll
    for (int j = 0; j < 4; ++j) {
      half16 b = load_fragB(Bt, din, (tg << 6) + (j << 4), kk << 5, lane);
      acc[j] = __builtin_amdgcn_wmma_f32_16x16x32_f16(false, a, false, b, (short)0,
                                                      acc[j], false, false);
    }
  }
  int rbase = (tm << 4) + ((lane >> 4) << 3);
#pragma unroll
  for (int j = 0; j < 4; ++j) {
    int n = (tg << 6) + (j << 4) + (lane & 15);
#pragma unroll
    for (int r = 0; r < 8; ++r) {
      int row = rbase + r;
      float v = acc[j][r];
      if (EPI == 0) {
        if (n < C_) {
          h2[(size_t)row * C_ + n] = v + bias[n];
        } else {
          int kidx = (n - C_) >> 6, c = (n - C_) & 63;
          g[(size_t)row * GCOLS + c * K_ + kidx] = (_Float16)v;   // swizzled
        }
      } else {
        g[(size_t)row * GCOLS + n] = (_Float16)fmaxf(v + bias[n], 0.f);
      }
    }
  }
}

// h2[row,c] += sum_k sum_n attn[k][m,n] * g_swz[b*M_+cols[m,n]][c*8+k]
// One 16B load per (thread, neighbor): 64 threads x 16B = contiguous 1KB per row.
__global__ void k_gather(const _Float16* __restrict__ g, const float* __restrict__ attn,
                         const int* __restrict__ L_idx, float* __restrict__ h2) {
  __shared__ float sA[K_ * NN_];
  __shared__ int   sC[NN_];
  int row = blockIdx.x;                // 0..ROWS-1
  int b = row >> 11;
  int m = row & (M_ - 1);
  int c = threadIdx.x;                 // 0..63
  for (int t = c; t < K_ * NN_; t += C_) {
    int k = t >> 5, n = t & 31;
    sA[t] = attn[(size_t)k * M_ * NN_ + (size_t)m * NN_ + n];
  }
  if (c < NN_) sC[c] = L_idx[m * NN_ + c] & (M_ - 1);   // cols = L_idx % M
  __syncthreads();
  float acc = 0.f;
#pragma unroll 4
  for (int n = 0; n < NN_; ++n) {
    const half8 v = *(const half8*)(g + (size_t)(b * M_ + sC[n]) * GCOLS + c * K_);
    if (n + 1 < NN_)
      __builtin_prefetch(g + (size_t)(b * M_ + sC[n + 1]) * GCOLS + c * K_, 0, 1);
#pragma unroll
    for (int k = 0; k < K_; ++k) acc += sA[k * NN_ + n] * (float)v[k];
  }
  h2[(size_t)row * C_ + c] += acc;
}

// zero the 128-float BN accumulator block
__global__ void k_zero(float* __restrict__ p) {
  if (threadIdx.x < 2 * C_) p[threadIdx.x] = 0.f;
}

// coalesced BN partial sums: thread t owns channel t&63; consecutive threads
// read consecutive addresses. 64 blocks x 256 threads -> 32768 float atomics.
__global__ void k_bnstat(const float* __restrict__ h2, float* __restrict__ acc) {
  int t = blockIdx.x * blockDim.x + threadIdx.x;      // 0..16383
  int c = t & 63;
  float s = 0.f, s2 = 0.f;
  for (size_t idx = t; idx < (size_t)ROWS * C_; idx += 16384) {
    float v = h2[idx];
    s += v; s2 += v * v;
  }
  atomicAdd(&acc[c], s);
  atomicAdd(&acc[C_ + c], s2);
}

__global__ void k_bnfin(const float* __restrict__ acc, float* __restrict__ meanv,
                        float* __restrict__ istdv) {
  int c = threadIdx.x;
  if (c >= C_) return;
  float mean = acc[c] / (float)ROWS;
  float var  = acc[C_ + c] / (float)ROWS - mean * mean;
  meanv[c] = mean;
  istdv[c] = rsqrtf(var + BN_EPS);
}

// h_next = relu( (h2-mean)*istd*gamma + beta ) as f16
__global__ void k_bnnorm(const float* __restrict__ h2, const float* __restrict__ meanv,
                         const float* __restrict__ istdv, const float* __restrict__ gamma,
                         const float* __restrict__ beta, _Float16* __restrict__ hout) {
  int t = blockIdx.x * blockDim.x + threadIdx.x;
  if (t >= ROWS * C_) return;
  int c = t & 63;
  float v = (h2[t] - meanv[c]) * istdv[c] * gamma[c] + beta[c];
  hout[t] = (_Float16)fmaxf(v, 0.f);
}

// out[row] = z[row,:] . fc2_w + fc2_b   (one wave per row)
__global__ void k_fc2(const _Float16* __restrict__ z, const float* __restrict__ w,
                      const float* __restrict__ bias, float* __restrict__ out) {
  int wave = (blockIdx.x * blockDim.x + threadIdx.x) >> 5;
  int lane = threadIdx.x & 31;
  if (wave >= ROWS) return;
  float acc = 0.f;
#pragma unroll
  for (int j = 0; j < GCOLS / 32; ++j) {
    int idx = lane + j * 32;
    acc += (float)z[(size_t)wave * GCOLS + idx] * w[idx];
  }
#pragma unroll
  for (int off = 16; off >= 1; off >>= 1) acc += __shfl_xor(acc, off, 32);
  if (lane == 0) out[wave] = acc + bias[0];
}

// ---------------- host orchestration ----------------
extern "C" void kernel_launch(void* const* d_in, const int* in_sizes, int n_in,
                              void* d_out, int out_size, void* d_ws, size_t ws_size,
                              hipStream_t stream) {
  (void)in_sizes; (void)n_in; (void)out_size; (void)ws_size;
  const float* x     = (const float*)d_in[0];
  const float* maps  = (const float*)d_in[1];
  const int*   L_idx = (const int*)d_in[2];
  const float *Wk[LAYERS_], *bk[LAYERS_], *W2[LAYERS_], *b2[LAYERS_];
  const float *sigma[LAYERS_], *mu[LAYERS_], *gamma[LAYERS_], *beta[LAYERS_];
  for (int i = 0; i < LAYERS_; ++i) {
    Wk[i]    = (const float*)d_in[3 + i];
    bk[i]    = (const float*)d_in[13 + i];
    W2[i]    = (const float*)d_in[23 + i];
    b2[i]    = (const float*)d_in[33 + i];
    sigma[i] = (const float*)d_in[43 + i];
    mu[i]    = (const float*)d_in[53 + i];
    gamma[i] = (const float*)d_in[63 + i];
    beta[i]  = (const float*)d_in[73 + i];
  }
  const float* fc1_w = (const float*)d_in[83];
  const float* fc1_b = (const float*)d_in[84];
  const float* fc2_w = (const float*)d_in[85];
  const float* fc2_b = (const float*)d_in[86];
  float* out = (float*)d_out;

  // workspace carve (256B aligned)
  char* ws = (char*)d_ws;
  auto carve = [&](size_t bytes) { char* p = ws; ws += (bytes + 255) & ~(size_t)255; return p; };
  _Float16* hf16  = (_Float16*)carve((size_t)ROWS * P_ * 2);        // 8 MB (max din=128)
  _Float16* g     = (_Float16*)carve((size_t)ROWS * GCOLS * 2);     // 33.5 MB
  float*    h2    = (float*)   carve((size_t)ROWS * C_ * 4);        // 8 MB
  float*    attn  = (float*)   carve((size_t)K_ * M_ * NN_ * 4);    // 2 MB
  _Float16* Wt    = (_Float16*)carve((size_t)NCOLS * P_ * 2);       // 144 KB
  float*    bsum  = (float*)   carve(C_ * 4);
  float*    bnacc = (float*)   carve(2 * C_ * 4);
  float*    meanv = (float*)   carve(C_ * 4);
  float*    istdv = (float*)   carve(C_ * 4);

  // x -> h f16 (tiled transpose)
  {
    dim3 grid(M_ / 32, P_ / 32, B_), blk(32, 8);
    k_transpose_x<<<grid, blk, 0, stream>>>(x, hf16);
  }

  for (int i = 0; i < LAYERS_; ++i) {
    int din = (i == 0) ? P_ : C_;
    k_attn<<<(K_ * M_) / 8, 256, 0, stream>>>(maps, sigma[i], mu[i], attn);
    int wcelems = NCOLS * din + C_;
    k_wconv<<<(wcelems + 255) / 256, 256, 0, stream>>>(W2[i], Wk[i], b2[i], bk[i],
                                                       Wt, bsum, din);
    // fused GEMM: (ROWS x din) @ (din x 576), 16x64 tile per wave
    int waves = (ROWS >> 4) * (NCOLS >> 6);                  // 2048*9 = 18432
    k_gemm_wmma<0><<<waves / 4, 128, 0, stream>>>(hf16, Wt, din, NCOLS, din / 32,
                                                  bsum, h2, g);
    k_gather<<<ROWS, C_, 0, stream>>>(g, attn, L_idx, h2);
    k_zero<<<1, 128, 0, stream>>>(bnacc);
    k_bnstat<<<C_, 256, 0, stream>>>(h2, bnacc);
    k_bnfin<<<1, C_, 0, stream>>>(bnacc, meanv, istdv);
    k_bnnorm<<<(ROWS * C_) / 256, 256, 0, stream>>>(h2, meanv, istdv,
                                                    gamma[i], beta[i], hf16);
  }

  // fc1: (ROWS x 64) @ (64 x 512), relu -> g (linear layout)
  k_wconv_fc1<<<(GCOLS * C_ + 255) / 256, 256, 0, stream>>>(fc1_w, Wt);
  int waves_fc = (ROWS >> 4) * (GCOLS >> 6);                 // 2048*8 = 16384
  k_gemm_wmma<1><<<waves_fc / 4, 128, 0, stream>>>(hf16, Wt, C_, GCOLS, C_ / 32,
                                                   fc1_b, h2, g);
  k_fc2<<<ROWS / 8, 256, 0, stream>>>(g, fc2_w, fc2_b, out);
}
